// stagenet_46445776339346
// MI455X (gfx1250) — matrix operand order repaired and checked by reading, not compiled, not run on hardware
//
#include <hip/hip_runtime.h>
#include <hip/hip_bf16.h>
#include <math.h>

// Problem constants (from reference setup_inputs)
#define V_  5
#define B_  2
#define C_  32
#define H_  128
#define W_  128
#define D_  32
#define G_  8
#define HW_ (H_ * W_)
#define NVIEW 4  // V_ - 1 source views

typedef __attribute__((ext_vector_type(16))) _Float16 v16h;
typedef __attribute__((ext_vector_type(8)))  float    v8f;

__device__ __forceinline__ int clampi(int v, int lo, int hi) {
    return v < lo ? lo : (v > hi ? hi : v);
}

// ---------------------------------------------------------------------------
// Setup: per (batch, src view) combined projection rot/trans.
// make_proj: top = K[:3,:3] @ E[:3,:4]; combined is affine [R t; 0 1].
// proj = comb_src @ inv(comb_ref) => rot = Rv*R0^-1, trans = tv - rot*t0.
// ---------------------------------------------------------------------------
__global__ void proj_setup_kernel(const float* __restrict__ projm,
                                  float* __restrict__ rt) {
    int t = threadIdx.x;
    if (t >= B_ * NVIEW) return;
    int b = t / NVIEW;
    int v = (t % NVIEW) + 1;

    float R0[9], t0[3], Rv[9], tv[3];
    for (int pass = 0; pass < 2; ++pass) {
        int view = pass == 0 ? 0 : v;
        const float* ext  = projm + ((size_t)(b * V_ + view) * 2 + 0) * 16;
        const float* intr = projm + ((size_t)(b * V_ + view) * 2 + 1) * 16;
        float* R  = pass == 0 ? R0 : Rv;
        float* tr = pass == 0 ? t0 : tv;
        for (int i = 0; i < 3; ++i) {
            for (int j = 0; j < 3; ++j) {
                float s = 0.f;
                for (int k = 0; k < 3; ++k) s += intr[i * 4 + k] * ext[k * 4 + j];
                R[i * 3 + j] = s;
            }
            float s = 0.f;
            for (int k = 0; k < 3; ++k) s += intr[i * 4 + k] * ext[k * 4 + 3];
            tr[i] = s;
        }
    }
    // 3x3 inverse of R0 via adjugate
    float c00 = R0[4] * R0[8] - R0[5] * R0[7];
    float c01 = R0[5] * R0[6] - R0[3] * R0[8];
    float c02 = R0[3] * R0[7] - R0[4] * R0[6];
    float id  = 1.0f / (R0[0] * c00 + R0[1] * c01 + R0[2] * c02);
    float inv[9];
    inv[0] = c00 * id;
    inv[1] = (R0[2] * R0[7] - R0[1] * R0[8]) * id;
    inv[2] = (R0[1] * R0[5] - R0[2] * R0[4]) * id;
    inv[3] = c01 * id;
    inv[4] = (R0[0] * R0[8] - R0[2] * R0[6]) * id;
    inv[5] = (R0[2] * R0[3] - R0[0] * R0[5]) * id;
    inv[6] = c02 * id;
    inv[7] = (R0[1] * R0[6] - R0[0] * R0[7]) * id;
    inv[8] = (R0[0] * R0[4] - R0[1] * R0[3]) * id;

    float rot[9], trs[3];
    for (int i = 0; i < 3; ++i)
        for (int j = 0; j < 3; ++j) {
            float s = 0.f;
            for (int k = 0; k < 3; ++k) s += Rv[i * 3 + k] * inv[k * 3 + j];
            rot[i * 3 + j] = s;
        }
    for (int i = 0; i < 3; ++i) {
        float s = 0.f;
        for (int k = 0; k < 3; ++k) s += rot[i * 3 + k] * t0[k];
        trs[i] = tv[i] - s;
    }
    float* o = rt + t * 12;
    for (int i = 0; i < 9; ++i) o[i] = rot[i];
    for (int i = 0; i < 3; ++i) o[9 + i] = trs[i];
}

// ---------------------------------------------------------------------------
// Main: one wave32 per pixel. Lane pair (n, n+16) serves depth n of a chunk:
// low lane gathers channels 0-15, high lane channels 16-31 (contiguous k order,
// used for both A and B: lanes 0-15 hold K=0-15, lanes 16-31 K=16-31).
// A holds ref features masked by group (rows 0-7), so
// v_wmma_f32_16x16x32_f16 computes 4*cor_feat[g, d] for 8 groups x 16 depths.
// ---------------------------------------------------------------------------
__global__ __launch_bounds__(256)
void stagenet_main(const float* __restrict__ feats,
                   const float* __restrict__ depth_hypo,
                   const float* __restrict__ reg_w,
                   const float* __restrict__ rt,
                   float* __restrict__ out_depth,   // [B,H,W]
                   float* __restrict__ out_attn) {  // [B,D,H,W]
    const int lane = threadIdx.x & 31;
    const int wave = threadIdx.x >> 5;
    const int pix  = blockIdx.x * 8 + wave;   // 0 .. B*H*W-1
    const int b  = pix / HW_;
    const int hw = pix % HW_;
    const int h  = hw / W_;
    const int w  = hw % W_;
    const bool hi = lane >= 16;
    const int  l16 = lane & 15;

    // ---- A fragment: grouped reference feature (16-bit A 16x32 layout) ----
    // lanes<16:  K=e (0..15)
    // lanes>=16: K=e+16 (16..31);  row M = lane & 15
    const float* refp = feats + (size_t)(0 * B_ + b) * C_ * HW_ + hw;
    const int m = l16;
    v16h afrag;
#pragma unroll
    for (int e = 0; e < 16; ++e) {
        int kA = e + (hi ? 16 : 0);
        float av = 0.0f;
        if (m < 8 && (kA >> 2) == m) av = refp[kA * HW_];
        afrag[e] = (_Float16)av;
    }

    const float inv_sqrt_c = 0.17677669529663687f;  // 1/sqrt(32)
    float cw_sum[2] = {1e-8f, 1e-8f};               // per my two depths (d, d+16)
    float cf[2][8];
#pragma unroll
    for (int q = 0; q < 2; ++q)
#pragma unroll
        for (int g = 0; g < 8; ++g) cf[q][g] = 0.f;

    const float fx = (float)w, fy = (float)h;

    for (int v = 0; v < NVIEW; ++v) {
        const float* R = rt + (b * NVIEW + v) * 12;
        const float rx = R[0] * fx + R[1] * fy + R[2];
        const float ry = R[3] * fx + R[4] * fy + R[5];
        const float rz = R[6] * fx + R[7] * fy + R[8];
        const float tx = R[9], ty = R[10], tz = R[11];
        const float* srcf = feats + (size_t)((v + 1) * B_ + b) * C_ * HW_;
        const float* sp = srcf + (hi ? 16 * HW_ : 0);

        v8f acc[2] = {v8f{}, v8f{}};
#pragma unroll
        for (int chunk = 0; chunk < 2; ++chunk) {
            const int d = l16 + chunk * 16;
            const float dep = depth_hypo[(size_t)(b * D_ + d) * HW_ + hw];
            float X = rx * dep + tx;
            float Y = ry * dep + ty;
            float Z = rz * dep + tz;
            Z = (Z == 0.f) ? 1e-9f : Z;
            const float px = X / Z, py = Y / Z;
            const float x0f = floorf(px), y0f = floorf(py);
            const int x0 = (int)x0f, y0 = (int)y0f;
            const int x1 = x0 + 1, y1 = y0 + 1;
            const float wx1 = px - x0f, wx0 = 1.f - wx1;
            const float wy1 = py - y0f, wy0 = 1.f - wy1;
            const bool vx0 = (x0 >= 0) & (x0 <= W_ - 1);
            const bool vx1 = (x1 >= 0) & (x1 <= W_ - 1);
            const bool vy0 = (y0 >= 0) & (y0 <= H_ - 1);
            const bool vy1 = (y1 >= 0) & (y1 <= H_ - 1);
            const float w00 = wx0 * wy0 * ((vx0 & vy0) ? 1.f : 0.f);
            const float w10 = wx1 * wy0 * ((vx1 & vy0) ? 1.f : 0.f);
            const float w01 = wx0 * wy1 * ((vx0 & vy1) ? 1.f : 0.f);
            const float w11 = wx1 * wy1 * ((vx1 & vy1) ? 1.f : 0.f);
            const int xc0 = clampi(x0, 0, W_ - 1), xc1 = clampi(x1, 0, W_ - 1);
            const int yc0 = clampi(y0, 0, H_ - 1), yc1 = clampi(y1, 0, H_ - 1);
            const int i00 = yc0 * W_ + xc0, i10 = yc0 * W_ + xc1;
            const int i01 = yc1 * W_ + xc0, i11 = yc1 * W_ + xc1;

            // B fragment: lane covers 16 channels for its depth (K = e or 16+e)
            v16h bfrag;
#pragma unroll
            for (int e = 0; e < 16; ++e) {
                const float* cp = sp + e * HW_;
                float val = w00 * cp[i00] + w10 * cp[i10] +
                            w01 * cp[i01] + w11 * cp[i11];
                bfrag[e] = (_Float16)val;
            }
            acc[chunk] = __builtin_amdgcn_wmma_f32_16x16x32_f16(
                false, afrag, false, bfrag, (short)0, acc[chunk], false, false);
        }

        // lane n<16 now holds acc[q][g] = 4*cor_feat[g, n + 16q]; hi lanes: 0.
        float t0 = 0.f, t1 = 0.f;
#pragma unroll
        for (int g = 0; g < 8; ++g) { t0 += acc[0][g]; t1 += acc[1][g]; }
        t0 *= 0.25f;  // = cor_feat.sum over g at depth n
        t1 *= 0.25f;  // = cor_feat.sum over g at depth n+16
        // softmax over 32 depths: halves reduce independently (xor masks < 16)
        float mx = fmaxf(t0, t1);
#pragma unroll
        for (int s = 8; s >= 1; s >>= 1) mx = fmaxf(mx, __shfl_xor(mx, s));
        const float e0 = __expf(t0 - mx), e1 = __expf(t1 - mx);
        float sum = e0 + e1;
#pragma unroll
        for (int s = 8; s >= 1; s >>= 1) sum += __shfl_xor(sum, s);
        const float cw0 = e0 / sum * inv_sqrt_c;
        const float cw1 = e1 / sum * inv_sqrt_c;
        cw_sum[0] += cw0;
        cw_sum[1] += cw1;
#pragma unroll
        for (int g = 0; g < 8; ++g) {
            cf[0][g] += cw0 * 0.25f * acc[0][g];
            cf[1][g] += cw1 * 0.25f * acc[1][g];
        }
    }

    // logits = reg_w . (cor_feats / cor_weight_sum)
    float logit0 = 0.f, logit1 = 0.f;
#pragma unroll
    for (int g = 0; g < 8; ++g) {
        const float rw = reg_w[g];
        logit0 += rw * (cf[0][g] / cw_sum[0]);
        logit1 += rw * (cf[1][g] / cw_sum[1]);
    }
    // attention softmax over depth
    float mx = fmaxf(logit0, logit1);
#pragma unroll
    for (int s = 8; s >= 1; s >>= 1) mx = fmaxf(mx, __shfl_xor(mx, s));
    const float e0 = __expf(logit0 - mx), e1 = __expf(logit1 - mx);
    float sum = e0 + e1;
#pragma unroll
    for (int s = 8; s >= 1; s >>= 1) sum += __shfl_xor(sum, s);
    const float a0 = e0 / sum, a1 = e1 / sum;

    if (!hi) {
        out_attn[(size_t)(b * D_ + l16) * HW_ + hw]        = a0;
        out_attn[(size_t)(b * D_ + l16 + 16) * HW_ + hw]   = a1;
    }

    // argmax over depth with first-max tie-breaking (smaller d wins)
    float bv;
    int bd;
    if (a0 >= a1) { bv = a0; bd = l16; } else { bv = a1; bd = l16 + 16; }
#pragma unroll
    for (int s = 8; s >= 1; s >>= 1) {
        const float ov = __shfl_xor(bv, s);
        const int   od = __shfl_xor(bd, s);
        if (ov > bv || (ov == bv && od < bd)) { bv = ov; bd = od; }
    }
    if (lane == 0) {
        out_depth[b * HW_ + hw] = depth_hypo[(size_t)(b * D_ + bd) * HW_ + hw];
    }
}

// ---------------------------------------------------------------------------
extern "C" void kernel_launch(void* const* d_in, const int* in_sizes, int n_in,
                              void* d_out, int out_size, void* d_ws, size_t ws_size,
                              hipStream_t stream) {
    const float* feats = (const float*)d_in[0];   // [V,B,C,H,W]
    const float* projm = (const float*)d_in[1];   // [B,V,2,4,4]
    const float* dh    = (const float*)d_in[2];   // [B,D,H,W]
    const float* rw    = (const float*)d_in[3];   // [G]
    (void)in_sizes; (void)n_in; (void)out_size; (void)ws_size;

    float* rt = (float*)d_ws;                     // 8 * 12 floats
    proj_setup_kernel<<<1, 32, 0, stream>>>(projm, rt);

    float* out_depth = (float*)d_out;             // B*H*W
    float* out_attn  = out_depth + B_ * HW_;      // B*D*H*W
    dim3 grid((B_ * HW_) / 8);                    // 8 waves (pixels) per block
    stagenet_main<<<grid, 256, 0, stream>>>(feats, dh, rw, rt, out_depth, out_attn);
}
